// SpikingMoEFFN_11897059410879
// MI455X (gfx1250) — matrix operand order repaired and checked by reading, not compiled, not run on hardware
//
#include <hip/hip_runtime.h>

// ---------------------------------------------------------------- constants
typedef unsigned int u32;
constexpr int kN = 16384;     // B*S tokens
constexpr int kD = 1024;      // model dim
constexpr int kH = 4096;      // hidden dim
constexpr int kE = 8;         // experts
constexpr int kMT = 16;       // tokens per M tile (WMMA M)
constexpr int kMTILES = kN / kMT;   // 1024 worst-case tiles per expert
constexpr int kHW32 = kH / 32;      // 128 bitmask words per hidden row
constexpr int kHCHUNK = 128;  // hidden cols per block in pass 2 (8 waves x 16)
constexpr int kDCHUNK = 128;  // out cols per block in pass 3   (8 waves x 16)

// ---------------------------------------------------------------- WMMA types
typedef __attribute__((ext_vector_type(16))) __bf16 v16bf;
typedef __attribute__((ext_vector_type(8)))  float  v8f;

union FragBF { v16bf v; u32 u[8]; };

__device__ __forceinline__ u32 f2bf(float f) {
  u32 b = __float_as_uint(f);
  return (b + 0x7FFFu + ((b >> 16) & 1u)) >> 16;   // round-to-nearest-even bf16
}

__device__ __forceinline__ float fast_rcp(float x) {
#if __has_builtin(__builtin_amdgcn_rcpf)
  return __builtin_amdgcn_rcpf(x);   // v_rcp_f32, no div chain
#else
  return 1.0f / x;
#endif
}

__device__ __forceinline__ v8f wmma_bf16(v16bf a, v16bf b, v8f c) {
  // (neg_a, A, neg_b, B, c_mod, C, reuse_a, reuse_b)
  return __builtin_amdgcn_wmma_f32_16x16x32_bf16(false, a, false, b,
                                                 (short)0, c, false, false);
}

// A fragment (16x32 bf16) from an LDS tile stored as u32[16][16]
// lane L: row = L%16; k = (L/16)*16 + 0..15
__device__ __forceinline__ v16bf afrag_lds(const u32 A[16][16], int lane) {
  const int m = lane & 15, half = lane >> 4;
  FragBF f;
#pragma unroll
  for (int j = 0; j < 4; ++j) {
    f.u[j]     = A[m][half * 8 + j];
    f.u[4 + j] = A[m][half * 8 + 4 + j];
  }
  return f.v;
}

// B fragment: 16 contiguous bf16 (8 dwords) straight from an LDS bf16 row
__device__ __forceinline__ v16bf bfrag_bf16(const u32* row16u, int half) {
  FragBF f;
#pragma unroll
  for (int j = 0; j < 8; ++j) f.u[j] = row16u[half * 8 + j];
  return f.v;
}

// B fragment: 16 contiguous fp32 from LDS, converted to bf16 (fallback path)
__device__ __forceinline__ v16bf bfrag_f32(const float* s) {
  FragBF f;
#pragma unroll
  for (int j = 0; j < 8; ++j)
    f.u[j] = f2bf(s[2 * j]) | (f2bf(s[2 * j + 1]) << 16);
  return f.v;
}

__device__ __forceinline__ u32 ballot32(bool p) {
#if __has_builtin(__builtin_amdgcn_ballot_w32)
  return __builtin_amdgcn_ballot_w32(p);
#else
  return (u32)__ballot(p);
#endif
}

// ------------------------------------------------- CDNA5 async paths
#ifndef ATHENA_TDM
#  if defined(__HIP_DEVICE_COMPILE__)
#    if __has_builtin(__builtin_amdgcn_tensor_load_to_lds) && \
        __has_builtin(__builtin_amdgcn_s_wait_tensorcnt)
#      define ATHENA_TDM 1
#    endif
#  endif
#  ifndef ATHENA_TDM
#    define ATHENA_TDM 0
#  endif
#endif

// direct global->LDS async DMA (ASYNCcnt) via inline asm
#ifndef ATHENA_ASYNC
#  if defined(__HIP_DEVICE_COMPILE__)
#    define ATHENA_ASYNC 1
#  else
#    define ATHENA_ASYNC 0
#  endif
#endif

__device__ __forceinline__ u32 lds_addr32(const void* p) {
  // flat shared address low 32 bits == LDS byte address (aperture in high bits)
  return (u32)(uintptr_t)p;
}

#if ATHENA_TDM
typedef __attribute__((ext_vector_type(4))) u32 v4u;
typedef __attribute__((ext_vector_type(8))) int v8i;
typedef __attribute__((ext_vector_type(4))) int v4i;

// 2D tile (tile_w elems x tile_h rows) from a row-major tensor -> LDS.
// dsize_code: 0=1B, 1=2B, 2=4B elements.
__device__ __forceinline__ void tdm_load_2d(u32 lds_byte_addr, const void* gp,
                                            u32 dsize_code,
                                            u32 tile_w, u32 tile_h,
                                            u32 tensor_w, u32 tensor_h,
                                            u32 row_stride_elems) {
  unsigned long long ga = (unsigned long long)(uintptr_t)gp;
  v4u g0 = {};
  g0[0] = 1u;                                         // count=1, user descriptor
  g0[1] = lds_byte_addr;                              // lds_addr
  g0[2] = (u32)ga;                                    // global_addr[31:0]
  g0[3] = ((u32)(ga >> 32) & 0x01FFFFFFu) | (2u << 30); // addr[56:32] | type=2
  v8i g1 = {};
  g1[0] = (int)(dsize_code << 16);                    // data_size
  g1[1] = (int)((tensor_w & 0xFFFFu) << 16);          // tensor_dim0 lo
  g1[2] = (int)((tensor_w >> 16) | ((tensor_h & 0xFFFFu) << 16));
  g1[3] = (int)(((tensor_h >> 16) & 0xFFFFu) | ((tile_w & 0xFFFFu) << 16));
  g1[4] = (int)(tile_h & 0xFFFFu);                    // tile_dim1 (tile_dim2=0)
  g1[5] = (int)row_stride_elems;                      // tensor_dim0_stride lo32
  g1[6] = 0;
  g1[7] = 0;
  v4i gz = {0, 0, 0, 0};
#if __clang_major__ >= 23
  v8i gz8 = {};
  __builtin_amdgcn_tensor_load_to_lds(g0, g1, gz, gz, gz8, 0);
#else
  __builtin_amdgcn_tensor_load_to_lds(g0, g1, gz, gz, 0);
#endif
}
#endif  // ATHENA_TDM

// ====================================================== pass 0: weight cvt
// one-time fp32 -> packed bf16x2; amortized across ~290 tile re-reads/expert
__global__ __launch_bounds__(256) void k_cvt_bf16(const float* __restrict__ src,
                                                  u32* __restrict__ dst,
                                                  int n2) {
  int i = blockIdx.x * 256 + threadIdx.x;
  if (i >= n2) return;
  float2 f = ((const float2*)src)[i];
  dst[i] = f2bf(f.x) | (f2bf(f.y) << 16);
}

// ================================================================ pass 1: gate
__global__ __launch_bounds__(256) void k_gate(
    const float* __restrict__ x, const float* __restrict__ Wg,
    const float* __restrict__ bg, unsigned short* __restrict__ xb,
    int* __restrict__ counts, int* __restrict__ idx, float* __restrict__ wts) {
  const int tokid = blockIdx.x * 8 + (threadIdx.x >> 5);
  const int lane = threadIdx.x & 31;
  if (tokid >= kN) return;
  const float* xr = x + (size_t)tokid * kD;
  float acc[kE] = {};
  for (int d = lane; d < kD; d += 32) {
    float xv = xr[d];
    xb[(size_t)tokid * kD + d] = (unsigned short)f2bf(xv);
#pragma unroll
    for (int e = 0; e < kE; ++e) acc[e] += xv * Wg[e * kD + d];
  }
#pragma unroll
  for (int e = 0; e < kE; ++e)
#pragma unroll
    for (int off = 16; off; off >>= 1) acc[e] += __shfl_xor(acc[e], off, 32);
  if (lane == 0) {
    u32 m = 0;
#pragma unroll
    for (int e = 0; e < kE; ++e)
      if (acc[e] + bg[e] > 1.0f) m |= 1u << e;
    int e0, e1;
    float w0, w1;
    if (m) {
      e0 = __ffs(m) - 1;
      u32 m2 = m & ~(1u << e0);
      if (m2) { e1 = __ffs(m2) - 1; w0 = w1 = 0.5f; }            // [1,1]
      else { e1 = (e0 == 0) ? 1 : 0;                              // [1,0]
             w0 = 0.731058579f; w1 = 0.268941421f; }
    } else { e0 = 0; e1 = 1; w0 = w1 = 0.5f; }                    // [0,0]
    int p0 = atomicAdd(&counts[e0], 1);
    idx[e0 * kN + p0] = tokid; wts[e0 * kN + p0] = w0;
    int p1 = atomicAdd(&counts[e1], 1);
    idx[e1 * kN + p1] = tokid; wts[e1 * kN + p1] = w1;
  }
}

// ====================================================== pass 2: hidden spikes
template <bool BF16W>
__global__ __launch_bounds__(256) void k_expert_hidden(
    const unsigned short* __restrict__ xb,
    const float* __restrict__ W1f, const float* __restrict__ W3f,
    const unsigned short* __restrict__ W1h,
    const unsigned short* __restrict__ W3h,
    const float* __restrict__ b1, const float* __restrict__ b3,
    const int* __restrict__ counts, const int* __restrict__ idx,
    u32* __restrict__ sbits) {
  const int mt = blockIdx.x, hc = blockIdx.y, e = blockIdx.z;
  const int cnt = counts[e];
  if (mt * kMT >= cnt) return;

  constexpr int TWu = BF16W ? (kHCHUNK * 16) : (kHCHUNK * 32);  // dwords/tile
  __shared__ __align__(16) u32 At[16][16];      // 16x32 bf16 A tile
  __shared__ __align__(16) u32 Wt[2][2][TWu];   // [matrix][buffer][tile]
  __shared__ int tok[16];
  __shared__ u32 spk[16][4];                    // per-row 128b spike mask

  const int tid = threadIdx.x, lane = tid & 31, w = tid >> 5;
  if (tid < 16) {
    int r = mt * kMT + tid;
    tok[tid] = (r < cnt) ? idx[e * kN + r] : 0;  // pad rows read token 0
  }
  if (tid < 64) spk[tid >> 2][tid & 3] = 0u;

  const int hbase = hc * kHCHUNK;
  const size_t wrow = ((size_t)e * kH + hbase) * kD;
  v8f accG = {}, accU = {};

#if ATHENA_TDM
  if (tid == 0) {  // prologue: first weight tile pair into buffer 0
    if constexpr (BF16W) {
      tdm_load_2d(lds_addr32(&Wt[0][0][0]), W1h + wrow, 1, 32, kHCHUNK,
                  (u32)kD, (u32)kH, (u32)kD);
      tdm_load_2d(lds_addr32(&Wt[1][0][0]), W3h + wrow, 1, 32, kHCHUNK,
                  (u32)kD, (u32)kH, (u32)kD);
    } else {
      tdm_load_2d(lds_addr32(&Wt[0][0][0]), W1f + wrow, 2, 32, kHCHUNK,
                  (u32)kD, (u32)kH, (u32)kD);
      tdm_load_2d(lds_addr32(&Wt[1][0][0]), W3f + wrow, 2, 32, kHCHUNK,
                  (u32)kD, (u32)kH, (u32)kD);
    }
  }
#endif

  for (int kb = 0; kb < kD; kb += 32) {
    const int cur = (kb >> 5) & 1;
    __syncthreads();  // previous K-step's fragment reads complete

    // ---- A tile: gathered token rows, direct async global->LDS DMA
#if ATHENA_ASYNC
    if (tid < 64) {
      int r = tid >> 2, c = tid & 3;
      u32 la = lds_addr32(&At[r][c * 4]);
      const void* ga = (const char*)(xb + (size_t)tok[r] * kD + kb) + c * 16;
      asm volatile("global_load_async_to_lds_b128 %0, %1, off"
                   :: "v"(la), "v"(ga) : "memory");
    }
#else
    if (tid < 64) {
      int r = tid >> 2, c = tid & 3;
      const uint4* src = (const uint4*)(xb + (size_t)tok[r] * kD + kb) + c;
      *(uint4*)&At[r][c * 4] = *src;
    }
#endif

    // ---- weight tiles: TDM double-buffer prefetch
#if ATHENA_TDM
    if (kb + 32 < kD) {
      if (tid == 0) {
        if constexpr (BF16W) {
          tdm_load_2d(lds_addr32(&Wt[0][cur ^ 1][0]), W1h + wrow + kb + 32, 1,
                      32, kHCHUNK, (u32)kD, (u32)kH, (u32)kD);
          tdm_load_2d(lds_addr32(&Wt[1][cur ^ 1][0]), W3h + wrow + kb + 32, 1,
                      32, kHCHUNK, (u32)kD, (u32)kH, (u32)kD);
        } else {
          tdm_load_2d(lds_addr32(&Wt[0][cur ^ 1][0]), W1f + wrow + kb + 32, 2,
                      32, kHCHUNK, (u32)kD, (u32)kH, (u32)kD);
          tdm_load_2d(lds_addr32(&Wt[1][cur ^ 1][0]), W3f + wrow + kb + 32, 2,
                      32, kHCHUNK, (u32)kD, (u32)kH, (u32)kD);
        }
      }
      // in-order TDM: <=2 outstanding => current pair done, next pair inflight
      __builtin_amdgcn_s_wait_tensorcnt(2);
    } else {
      __builtin_amdgcn_s_wait_tensorcnt(0);
    }
#else
    if constexpr (BF16W) {
      for (int t = tid; t < kHCHUNK * 16; t += 256) {
        int r = t >> 4, c = t & 15;
        Wt[0][cur][r * 16 + c] =
            ((const u32*)(W1h + wrow + (size_t)r * kD + kb))[c];
        Wt[1][cur][r * 16 + c] =
            ((const u32*)(W3h + wrow + (size_t)r * kD + kb))[c];
      }
    } else {
      for (int t = tid; t < kHCHUNK * 32; t += 256) {
        int r = t >> 5, c = t & 31;
        ((float*)&Wt[0][cur][0])[r * 32 + c] = W1f[wrow + (size_t)r * kD + kb + c];
        ((float*)&Wt[1][cur][0])[r * 32 + c] = W3f[wrow + (size_t)r * kD + kb + c];
      }
    }
#endif
#if ATHENA_ASYNC
    asm volatile("s_wait_asynccnt 0" ::: "memory");
#endif
    __syncthreads();

    const int n = lane & 15, half = lane >> 4;
    v16bf a = afrag_lds(At, lane);
    v16bf bG, bU;
    if constexpr (BF16W) {
      bG = bfrag_bf16(&Wt[0][cur][(w * 16 + n) * 16], half);
      bU = bfrag_bf16(&Wt[1][cur][(w * 16 + n) * 16], half);
    } else {
      bG = bfrag_f32((const float*)&Wt[0][cur][0] + (w * 16 + n) * 32 + half * 16);
      bU = bfrag_f32((const float*)&Wt[1][cur][0] + (w * 16 + n) * 32 + half * 16);
    }
    accG = wmma_bf16(a, bG, accG);
    accU = wmma_bf16(a, bU, accU);
  }

  // bias + SiLU*up, threshold, pack into bitmask (ballot + ds_or)
  const int n = lane & 15;
  const int h = hbase + w * 16 + n;
  const float bb1 = b1[e * kH + h], bb3 = b3[e * kH + h];
  const int wword = w >> 1, wsh = (w & 1) * 16;
#pragma unroll
  for (int i = 0; i < 8; ++i) {
    float g = accG[i] + bb1;
    float u = accU[i] + bb3;
    float hv = g * fast_rcp(1.0f + __expf(-g)) * u;   // silu(g) * up
    u32 msk = ballot32(hv > 1.0f);  // lanes 0-15: row i, lanes 16-31: row i+8
    if (lane == 0) {
      atomicOr(&spk[i][wword], (msk & 0xFFFFu) << wsh);
      atomicOr(&spk[i + 8][wword], (msk >> 16) << wsh);
    }
  }
  __syncthreads();
  if (tid < 64) {
    int m = tid >> 2, wd = tid & 3;
    sbits[(((size_t)e * kMTILES + mt) * kMT + m) * kHW32 + hc * 4 + wd] =
        spk[m][wd];
  }
}

// ================================================= pass 3: Y = spikes @ W2^T
template <bool BF16W>
__global__ __launch_bounds__(256) void k_expert_out(
    const u32* __restrict__ sbits, const float* __restrict__ W2f,
    const unsigned short* __restrict__ W2h, const float* __restrict__ b2,
    const int* __restrict__ counts, const int* __restrict__ idx,
    const float* __restrict__ wts, float* __restrict__ out) {
  const int mt = blockIdx.x, dc = blockIdx.y, e = blockIdx.z;
  const int cnt = counts[e];
  if (mt * kMT >= cnt) return;
  const int rows = (cnt - mt * kMT < kMT) ? (cnt - mt * kMT) : kMT;

  constexpr int TWu = BF16W ? (kDCHUNK * 16) : (kDCHUNK * 32);
  __shared__ __align__(16) u32 W2s[2][TWu];
  __shared__ int tok[16];
  __shared__ float twt[16];

  const int tid = threadIdx.x, lane = tid & 31, w = tid >> 5;
  if (tid < 16) {
    int r = mt * kMT + tid;
    bool v = r < cnt;
    tok[tid] = v ? idx[e * kN + r] : 0;
    twt[tid] = v ? wts[e * kN + r] : 0.0f;
  }

  const int dbase = dc * kDCHUNK;
  const size_t w2row = ((size_t)e * kD + dbase) * kH;
  const int n = lane & 15, half = lane >> 4;
  const u32* srow = sbits + (((size_t)e * kMTILES + mt) * kMT + n) * kHW32;
  v8f acc = {};

#if ATHENA_TDM
  if (tid == 0) {
    if constexpr (BF16W)
      tdm_load_2d(lds_addr32(&W2s[0][0]), W2h + w2row, 1, 32, kDCHUNK,
                  (u32)kH, (u32)kD, (u32)kH);
    else
      tdm_load_2d(lds_addr32(&W2s[0][0]), W2f + w2row, 2, 32, kDCHUNK,
                  (u32)kH, (u32)kD, (u32)kH);
  }
#endif

  for (int kb = 0; kb < kH; kb += 32) {
    const int cur = (kb >> 5) & 1;
    __syncthreads();
#if ATHENA_TDM
    if (kb + 32 < kH) {
      if (tid == 0) {
        if constexpr (BF16W)
          tdm_load_2d(lds_addr32(&W2s[cur ^ 1][0]), W2h + w2row + kb + 32, 1,
                      32, kDCHUNK, (u32)kH, (u32)kD, (u32)kH);
        else
          tdm_load_2d(lds_addr32(&W2s[cur ^ 1][0]), W2f + w2row + kb + 32, 2,
                      32, kDCHUNK, (u32)kH, (u32)kD, (u32)kH);
      }
      __builtin_amdgcn_s_wait_tensorcnt(1);  // current tile done, next inflight
    } else {
      __builtin_amdgcn_s_wait_tensorcnt(0);
    }
#else
    if constexpr (BF16W) {
      for (int t = tid; t < kDCHUNK * 16; t += 256) {
        int r = t >> 4, c = t & 15;
        W2s[cur][r * 16 + c] =
            ((const u32*)(W2h + w2row + (size_t)r * kH + kb))[c];
      }
    } else {
      for (int t = tid; t < kDCHUNK * 32; t += 256) {
        int r = t >> 5, c = t & 31;
        ((float*)&W2s[cur][0])[r * 32 + c] = W2f[w2row + (size_t)r * kH + kb + c];
      }
    }
#endif
    __syncthreads();

    u32 word = srow[kb >> 5];  // 32 hidden bits for this lane's token row
    FragBF fa;
#pragma unroll
    for (int j = 0; j < 4; ++j) {
      u32 p0 = (word >> (half * 16 + 2 * j)) & 1u;
      u32 p1 = (word >> (half * 16 + 2 * j + 1)) & 1u;
      fa.u[j] = (p0 ? 0x3F80u : 0u) | (p1 ? 0x3F800000u : 0u);
      u32 q0 = (word >> (half * 16 + 8 + 2 * j)) & 1u;
      u32 q1 = (word >> (half * 16 + 8 + 2 * j + 1)) & 1u;
      fa.u[4 + j] = (q0 ? 0x3F80u : 0u) | (q1 ? 0x3F800000u : 0u);
    }
    v16bf bw;
    if constexpr (BF16W)
      bw = bfrag_bf16(&W2s[cur][(w * 16 + n) * 16], half);
    else
      bw = bfrag_f32((const float*)&W2s[cur][0] + (w * 16 + n) * 32 + half * 16);
    acc = wmma_bf16(fa.v, bw, acc);
  }

  const int d = dbase + w * 16 + n;
  const float bb2 = b2[e * kD + d];
#pragma unroll
  for (int i = 0; i < 8; ++i) {
    int m = i + half * 8;  // C-layout row for this VGPR/lane
    if (m < rows)
      unsafeAtomicAdd(&out[(size_t)tok[m] * kD + d], twt[m] * (acc[i] + bb2));
  }
}

// ================================================================== launcher
extern "C" void kernel_launch(void* const* d_in, const int* in_sizes, int n_in,
                              void* d_out, int out_size, void* d_ws,
                              size_t ws_size, hipStream_t stream) {
  (void)in_sizes; (void)n_in; (void)out_size;
  const float* x  = (const float*)d_in[0];
  const float* Wg = (const float*)d_in[1];
  const float* bg = (const float*)d_in[2];
  const float* W1 = (const float*)d_in[3];
  const float* b1 = (const float*)d_in[4];
  const float* W3 = (const float*)d_in[5];
  const float* b3 = (const float*)d_in[6];
  const float* W2 = (const float*)d_in[7];
  const float* b2 = (const float*)d_in[8];
  float* out = (float*)d_out;

  // workspace carve-up: 102 MB base + 201 MB optional bf16 weight cache
  size_t off = 0;
  char* base = (char*)d_ws;
  auto carve = [&](size_t bytes) { char* p = base + off; off += bytes; return p; };
  unsigned short* xb = (unsigned short*)carve((size_t)kN * kD * 2);
  int* idx    = (int*)carve((size_t)kE * kN * 4);
  float* wts  = (float*)carve((size_t)kE * kN * 4);
  int* counts = (int*)carve(256);
  u32* sbits  = (u32*)carve((size_t)kE * kMTILES * kMT * kHW32 * 4);
  const size_t wElems = (size_t)kE * kH * kD;   // identical for W1/W3/W2
  unsigned short* w1h = (unsigned short*)carve(wElems * 2);
  unsigned short* w3h = (unsigned short*)carve(wElems * 2);
  unsigned short* w2h = (unsigned short*)carve(wElems * 2);
  const bool useBF16W = ws_size >= off;   // fixed per process -> deterministic

  hipMemsetAsync(counts, 0, 256, stream);
  hipMemsetAsync(out, 0, (size_t)kN * kD * sizeof(float), stream);

  k_gate<<<dim3(kN / 8), 256, 0, stream>>>(x, Wg, bg, xb, counts, idx, wts);

  if (useBF16W) {
    const int n2 = (int)(wElems / 2);
    k_cvt_bf16<<<dim3(n2 / 256), 256, 0, stream>>>(W1, (u32*)w1h, n2);
    k_cvt_bf16<<<dim3(n2 / 256), 256, 0, stream>>>(W3, (u32*)w3h, n2);
    k_cvt_bf16<<<dim3(n2 / 256), 256, 0, stream>>>(W2, (u32*)w2h, n2);
    k_expert_hidden<true><<<dim3(kMTILES, kH / kHCHUNK, kE), 256, 0, stream>>>(
        xb, W1, W3, w1h, w3h, b1, b3, counts, idx, sbits);
    k_expert_out<true><<<dim3(kMTILES, kD / kDCHUNK, kE), 256, 0, stream>>>(
        sbits, W2, w2h, b2, counts, idx, wts, out);
  } else {
    k_expert_hidden<false><<<dim3(kMTILES, kH / kHCHUNK, kE), 256, 0, stream>>>(
        xb, W1, W3, w1h, w3h, b1, b3, counts, idx, sbits);
    k_expert_out<false><<<dim3(kMTILES, kD / kDCHUNK, kE), 256, 0, stream>>>(
        sbits, W2, w2h, b2, counts, idx, wts, out);
  }
}
